// TrajectoryEncoder_28415503630418
// MI455X (gfx1250) — matrix operand order repaired and checked by reading, not compiled, not run on hardware
//
#include <hip/hip_runtime.h>

typedef __attribute__((ext_vector_type(16))) _Float16 v16h;
typedef __attribute__((ext_vector_type(8)))  _Float16 v8h;
typedef __attribute__((ext_vector_type(8)))  float    v8f;

#define Bq 8
#define Tq 200
#define Nq 512
#define Dq 4
#define Hq 64
#define Lq 32
#define HPAD 72   // padded LDS row stride in halves (16 rows x 72 halves)

#define L2E  1.4426950408889634f   // log2(e)

__device__ __forceinline__ float fast_rcp(float x) {
    return __builtin_amdgcn_rcpf(x);        // v_rcp_f32 (fast, ~1 ulp)
}
__device__ __forceinline__ float fast_exp2(float x) {
#if __has_builtin(__builtin_amdgcn_exp2f)
    return __builtin_amdgcn_exp2f(x);       // v_exp_f32, no pre-scale
#else
    return exp2f(x);
#endif
}

__device__ __forceinline__ v8f wmma32(v16h a, v16h b, v8f c) {
    // D = A(16x32 f16) x B(32x16 f16) + C(16x16 f32)
    return __builtin_amdgcn_wmma_f32_16x16x32_f16(
        /*neg_a=*/false, a, /*neg_b=*/false, b,
        /*c_mod=*/(short)0, c, /*reuse_a=*/false, /*reuse_b=*/false);
}

// B-operand fragment (32 K x 16 N) from a row-major f32 weight matrix with
// row length ROWLEN. Lane's column = `row` index; lanes 0-15 carry K {k0..k0+7, k0+16..k0+23},
// lanes 16-31 carry K {k0+8..k0+15, k0+24..k0+31} (matches load_a).
template<int ROWLEN>
__device__ __forceinline__ v16h load_b_chunk(const float* __restrict__ W,
                                             int row, int k0, bool hi) {
    const float* p = W + row * ROWLEN + k0 + (hi ? 8 : 0);
    v16h r;
#pragma unroll
    for (int i = 0; i < 4; ++i) {
        float4 f = reinterpret_cast<const float4*>(p + (i >= 2 ? 16 : 0))[i & 1];
        r[4 * i + 0] = (_Float16)f.x;
        r[4 * i + 1] = (_Float16)f.y;
        r[4 * i + 2] = (_Float16)f.z;
        r[4 * i + 3] = (_Float16)f.w;
    }
    return r;
}

// B-operand fragment for the x-contribution: K=0..3 real (W_ih row, D=4),
// K=4..31 zero.
__device__ __forceinline__ v16h load_bx_chunk(const float* __restrict__ W_ih,
                                              int row, bool hi) {
    v16h r = {};
    if (!hi) {
        float4 f = *reinterpret_cast<const float4*>(W_ih + row * Dq);
        r[0] = (_Float16)f.x; r[1] = (_Float16)f.y;
        r[2] = (_Float16)f.z; r[3] = (_Float16)f.w;
    }
    return r;
}

// A-operand fragment (16 M x 32 K f16) from row-major f16 LDS tile (stride HPAD).
// ISA A layout: lane l<16 holds row l, K {k0..k0+7, k0+16..k0+23};
// lane l+16 holds row l, K {k0+8..k0+15, k0+24..k0+31}.
__device__ __forceinline__ v16h load_a(const _Float16* hb, int l, bool hi, int k0) {
    const _Float16* p = hb + l * HPAD + k0 + (hi ? 8 : 0);
    v8h lo = *reinterpret_cast<const v8h*>(p);
    v8h hp = *reinterpret_cast<const v8h*>(p + 16);
    return __builtin_shufflevector(lo, hp, 0, 1, 2, 3, 4, 5, 6, 7,
                                   8, 9, 10, 11, 12, 13, 14, 15);
}

__global__ __launch_bounds__(128)
void gru_enc_kernel(const float* __restrict__ x,      // (B,T,N,D)
                    const float* __restrict__ W_ih,   // (192,4)
                    const float* __restrict__ W_hh,   // (192,64)
                    const float* __restrict__ b_ih,   // (192)
                    const float* __restrict__ b_hh,   // (192)
                    const float* __restrict__ W_enc,  // (32,64)
                    const float* __restrict__ b_enc,  // (32)
                    float* __restrict__ out)          // (B,N,32) == (M,32)
{
    __shared__ _Float16 hbuf0[16][HPAD];   // double-buffered h (16 seq x 64 hid, f16)
    __shared__ _Float16 hbuf1[16][HPAD];

    const int tid  = threadIdx.x;
    const int lane = tid & 31;
    const int wave = tid >> 5;        // 0..3: this wave owns hidden cols [16w,16w+16)
    const int l    = lane & 15;
    const bool hi  = lane >= 16;

    // Gate-row indices (columns of the fused 96x192 weight) for this wave/lane.
    const int colR = 0   + wave * 16 + l;
    const int colZ = 64  + wave * 16 + l;
    const int colN = 128 + wave * 16 + l;

    // ---- Loop-invariant weight B-operands: stay in VGPRs for all 200 steps ----
    v16h br0 = load_b_chunk<Hq>(W_hh, colR, 0,  hi);
    v16h br1 = load_b_chunk<Hq>(W_hh, colR, 32, hi);
    v16h brx = load_bx_chunk(W_ih, colR, hi);
    v16h bz0 = load_b_chunk<Hq>(W_hh, colZ, 0,  hi);
    v16h bz1 = load_b_chunk<Hq>(W_hh, colZ, 32, hi);
    v16h bzx = load_bx_chunk(W_ih, colZ, hi);
    v16h bn0 = load_b_chunk<Hq>(W_hh, colN, 0,  hi);
    v16h bn1 = load_b_chunk<Hq>(W_hh, colN, 32, hi);
    v16h bnx = load_bx_chunk(W_ih, colN, hi);

    // Pre-scaled sigmoid constants: sigmoid(a+b) = rcp(1 + exp2(fma(a,-L2E,c)))
    const float cR   = -L2E * (b_ih[colR] + b_hh[colR]);
    const float cZ   = -L2E * (b_ih[colZ] + b_hh[colZ]);
    const float bihN = b_ih[colN];   // stays outside r*(...)
    const float bhhN = b_hh[colN];   // stays inside  r*(...)

    // ---- Hidden state (f32, C-layout): VGPR v <-> seq row v / v+8 ----
    float h[8];
#pragma unroll
    for (int v = 0; v < 8; ++v) h[v] = 0.0f;

    const int row0 = hi ? 8 : 0;
    const int col  = wave * 16 + l;

    // publish h0 = 0 into buffer 0 (4 waves together cover all 64 columns)
#pragma unroll
    for (int v = 0; v < 8; ++v) hbuf0[row0 + v][col] = (_Float16)0.0f;
    __syncthreads();

    // x addressing: sequence m = blockIdx.x*16 + l ; x[(b*T+t)*N + n][d]
    const int m  = blockIdx.x * 16 + l;
    const int bb = m >> 9;          // / 512
    const int nn = m & 511;
    const float* xbase = x + ((size_t)bb * Tq * Nq + nn) * Dq;

    // One GRU step: read h from `hc`, write h(t+1) into `hnx`.
    auto step = [&](const _Float16 (*hc)[HPAD], _Float16 (*hnx)[HPAD], int t) {
        // A-operands: full 16x64 hidden tile (shared by all 4 waves via LDS)
        v16h ah0 = load_a(&hc[0][0], l, hi, 0);
        v16h ah1 = load_a(&hc[0][0], l, hi, 32);

        // x_t for this lane's sequence (one b128), prefetch ahead
        float4 xi = *reinterpret_cast<const float4*>(xbase + (size_t)t * (Nq * Dq));
        if (t + 4 < Tq)
            __builtin_prefetch(xbase + (size_t)(t + 4) * (Nq * Dq), 0, 1);

        v16h ax = {};   // A-operand, K=0..3 real x values (lanes 0-15), rest zero
        if (!hi) {
            ax[0] = (_Float16)xi.x; ax[1] = (_Float16)xi.y;
            ax[2] = (_Float16)xi.z; ax[3] = (_Float16)xi.w;
        }

        // Fused K=96 matmuls: gh+gx for r and z; gh and gx kept separate for n
        v8f accR = {};
        accR = wmma32(ah0, br0, accR);
        accR = wmma32(ah1, br1, accR);
        accR = wmma32(ax,  brx, accR);
        v8f accZ = {};
        accZ = wmma32(ah0, bz0, accZ);
        accZ = wmma32(ah1, bz1, accZ);
        accZ = wmma32(ax,  bzx, accZ);
        v8f accN = {};                       // gh_n (h-part only)
        accN = wmma32(ah0, bn0, accN);
        accN = wmma32(ah1, bn1, accN);
        v8f accX = {};                       // gx_n (x-part only)
        accX = wmma32(ax, bnx, accX);

        // Elementwise GRU update for this wave's 16 hidden columns.
        // sigmoid: 1 fma + 1 exp + 1 add + 1 rcp (bias & log2e pre-folded)
#pragma unroll
        for (int v = 0; v < 8; ++v) {
            float r  = fast_rcp(1.0f + fast_exp2(__builtin_fmaf(accR[v], -L2E, cR)));
            float z  = fast_rcp(1.0f + fast_exp2(__builtin_fmaf(accZ[v], -L2E, cZ)));
            float s  = accN[v] + bhhN;
            float w  = accX[v] + bihN;
            float u  = __builtin_fmaf(r, s, w);              // tanh argument
            float q  = fast_rcp(1.0f + fast_exp2(u * (2.0f * L2E)));
            float nv = __builtin_fmaf(-2.0f, q, 1.0f);       // tanh(u)
            h[v] = __builtin_fmaf(z, h[v] - nv, nv);         // (1-z)*n + z*h
            hnx[row0 + v][col] = (_Float16)h[v];             // publish for next step
        }
        __syncthreads();
    };

    // Tq is even: unroll x2 so buffer pointers are compile-time constant.
    for (int t = 0; t < Tq; t += 2) {
        step(hbuf0, hbuf1, t);
        step(hbuf1, hbuf0, t + 1);
    }

    // ---- Final projection: out = hT @ W_enc^T + b_enc (L=32 -> 2 tiles) ----
    if (wave < 2) {
        v16h ah0 = load_a(&hbuf0[0][0], l, hi, 0);
        v16h ah1 = load_a(&hbuf0[0][0], l, hi, 32);
        const int lo = wave * 16 + l;               // output feature
        v16h be0 = load_b_chunk<Hq>(W_enc, lo, 0,  hi);
        v16h be1 = load_b_chunk<Hq>(W_enc, lo, 32, hi);
        v8f acc = {};
        acc = wmma32(ah0, be0, acc);
        acc = wmma32(ah1, be1, acc);
        const float be = b_enc[lo];
#pragma unroll
        for (int v = 0; v < 8; ++v) {
            out[(size_t)(blockIdx.x * 16 + row0 + v) * Lq + lo] = acc[v] + be;
        }
    }
}

extern "C" void kernel_launch(void* const* d_in, const int* in_sizes, int n_in,
                              void* d_out, int out_size, void* d_ws, size_t ws_size,
                              hipStream_t stream) {
    const float* x     = (const float*)d_in[0];
    const float* W_ih  = (const float*)d_in[1];
    const float* W_hh  = (const float*)d_in[2];
    const float* b_ih  = (const float*)d_in[3];
    const float* b_hh  = (const float*)d_in[4];
    const float* W_enc = (const float*)d_in[5];
    const float* b_enc = (const float*)d_in[6];
    float* out = (float*)d_out;

    dim3 grid((Bq * Nq) / 16);   // 256 workgroups: one 16-sequence tile each
    dim3 block(128);             // 4 waves: split the 3H=192 gate columns
    hipLaunchKernelGGL(gru_enc_kernel, grid, block, 0, stream,
                       x, W_ih, W_hh, b_ih, b_hh, W_enc, b_enc, out);
}
